// MultiHeadSelfAttention_37546604101898
// MI455X (gfx1250) — matrix operand order repaired and checked by reading, not compiled, not run on hardware
//
#include <hip/hip_runtime.h>

// MI455X / gfx1250 multi-head self-attention, exploiting the reference's
// degenerate einsums:
//   hx/hWq/hWk/hWo = f16 copies (one pass)
//   Qh = hx @ hWq.T, Kh = hx @ hWk.T                  [WMMA GEMM, f32 acc]
//   WvS[d,:] = sum_h Wv[h*64+d,:]  (pre-summed V weight, 64x1024, f16)
//   Vs = hx @ WvS.T                                    [WMMA GEMM]
//   energy[n,qh,kh,s] = sum_d Qh[tok,qh*64+d]*Kh[tok,kh*64+d]  [WMMA/token]
//   per-(n,qh,kh) softmax stats over s; A[n,qh,s] = sum_kh softmax(...)
//   Y[n, qh*256+s/16, (s%16)*64+d] = f16(A[n,qh,s]*Vs[n,s,d])
//   out = Y @ hWo.T + bo                               [WMMA GEMM, bias]
// GEMM LDS staging uses GLOBAL_LOAD_ASYNC_TO_LDS_B128 with double-buffered
// tiles and s_wait_asynccnt pipelining when the toolchain exposes it.

typedef __attribute__((ext_vector_type(16))) _Float16 v16h;
typedef __attribute__((ext_vector_type(8)))  _Float16 v8h;
typedef __attribute__((ext_vector_type(8)))  float    v8f;
typedef int v4i __attribute__((vector_size(4 * sizeof(int))));

#define Nb 4
#define Sq 4096
#define Ed 1024
#define Hh 16
#define Dd 64

// ---- async copy to LDS (CDNA5), with safe fallback --------------------------
#if defined(__has_builtin)
#if __has_builtin(__builtin_amdgcn_global_load_async_to_lds_b128)
#define HAVE_ASYNC_LDS 1
#endif
#endif
#ifndef HAVE_ASYNC_LDS
#define HAVE_ASYNC_LDS 0
#endif

#if HAVE_ASYNC_LDS
#if defined(__has_builtin)
#if __has_builtin(__builtin_amdgcn_s_wait_asynccnt)
#define WAIT_ASYNC(n) __builtin_amdgcn_s_wait_asynccnt(n)
#endif
#endif
#ifndef WAIT_ASYNC
#define WAIT_ASYNC(n) asm volatile("s_wait_asynccnt " #n ::: "memory")
#endif
__device__ inline void async_cp16(const _Float16* g, _Float16* l) {
    __builtin_amdgcn_global_load_async_to_lds_b128(
        (__attribute__((address_space(1))) v4i*)(g),
        (__attribute__((address_space(3))) v4i*)(l), 0, 0);
}
#else
#define WAIT_ASYNC(n)
__device__ inline void async_cp16(const _Float16* g, _Float16* l) {
    *(v8h*)l = *(const v8h*)g;   // synchronous fallback, same semantics
}
#endif

__device__ inline v16h make_frag(const _Float16* p0, const _Float16* p1) {
    union { v16h v; v8h h[2]; } u;
    u.h[0] = *(const v8h*)p0;
    u.h[1] = *(const v8h*)p1;
    return u.v;
}

// ---------------------------------------------------------------------------
// f32 -> f16 bulk convert (8 elems/thread)
// ---------------------------------------------------------------------------
__global__ __launch_bounds__(256) void f32_to_f16_kernel(
    const float* __restrict__ src, _Float16* __restrict__ dst)
{
    int i = blockIdx.x * 256 + threadIdx.x;          // handles 8 elements
    const float4* s4 = (const float4*)src + (size_t)i * 2;
    float4 a = s4[0], b = s4[1];
    v8h o = { (_Float16)a.x, (_Float16)a.y, (_Float16)a.z, (_Float16)a.w,
              (_Float16)b.x, (_Float16)b.y, (_Float16)b.z, (_Float16)b.w };
    *((v8h*)dst + i) = o;
}

// ---------------------------------------------------------------------------
// Generic f16 WMMA GEMM: C[M,Nout] = A[M,K] @ B[Nout,K]^T (+ bias), fp32 acc.
// Block tile 128x64, 8 waves (4x2), each wave 32x32 (2x2 WMMA tiles), BLK_K=32.
// Double-buffered LDS tiles filled via async-to-LDS copies (ASYNCcnt).
// ---------------------------------------------------------------------------
template <typename TC, bool BIAS>
__global__ __launch_bounds__(256) void wmma_gemm_kernel(
    const _Float16* __restrict__ A, const _Float16* __restrict__ B,
    TC* __restrict__ C, const float* __restrict__ bias,
    int M, int Nout, int K)
{
    __shared__ _Float16 lA[2][128][40];   // padded rows (80B) to spread banks
    __shared__ _Float16 lB[2][64][40];

    const int tid  = threadIdx.x;
    const int lane = tid & 31;
    const int wave = tid >> 5;
    const int mBase = blockIdx.x * 128;
    const int nBase = blockIdx.y * 64;
    const int wm = (wave & 3) * 32;       // wave tile row within block tile
    const int wn = (wave >> 2) * 32;      // wave tile col within block tile
    const int fr  = lane & 15;
    const int kgA = (lane >> 4) * 8;      // A frag: chunks at kgA, kgA+16
    const int kgB = (lane >> 4) * 8;      // B frag: chunks at kgB, kgB+16

    // stage one 32-wide K slab of A (128x32) and B (64x32) into buffer b
    auto stage = [&](int k0, int b) {
        #pragma unroll
        for (int i = 0; i < 2; ++i) {                 // A: 512 x 16B chunks
            int flat = i * 256 + tid;
            int r = flat >> 2, c = (flat & 3) * 8;
            async_cp16(A + (size_t)(mBase + r) * K + k0 + c, &lA[b][r][c]);
        }
        {                                             // B: 256 x 16B chunks
            int r = tid >> 2, c = (tid & 3) * 8;
            async_cp16(B + (size_t)(nBase + r) * K + k0 + c, &lB[b][r][c]);
        }
    };

    v8f acc[2][2] = {};
    const int nIter = K >> 5;

    stage(0, 0);
    for (int it = 0; it < nIter; ++it) {
        const int bc = it & 1;
        if (it + 1 < nIter) {
            stage((it + 1) << 5, bc ^ 1);   // overlap next tile's copies
            WAIT_ASYNC(3);                  // retire current tile (in-order)
        } else {
            WAIT_ASYNC(0);
        }
        __syncthreads();

        v16h aF[2], bF[2];
        #pragma unroll
        for (int mi = 0; mi < 2; ++mi) {
            const _Float16* p = &lA[bc][wm + mi * 16 + fr][0];
            aF[mi] = make_frag(p + kgA, p + kgA + 16);
        }
        #pragma unroll
        for (int ni = 0; ni < 2; ++ni) {
            const _Float16* p = &lB[bc][wn + ni * 16 + fr][0];
            bF[ni] = make_frag(p + kgB, p + kgB + 16);
        }
        #pragma unroll
        for (int mi = 0; mi < 2; ++mi)
            #pragma unroll
            for (int ni = 0; ni < 2; ++ni)
                acc[mi][ni] = __builtin_amdgcn_wmma_f32_16x16x32_f16(
                    false, aF[mi], false, bF[ni], (short)0, acc[mi][ni],
                    false, false);
        __syncthreads();
    }

    // epilogue: C layout  M = r + 8*(lane/16), N = lane%16
    const int rowOff = (lane >> 4) * 8;
    const int col    = lane & 15;
    #pragma unroll
    for (int mi = 0; mi < 2; ++mi)
        #pragma unroll
        for (int ni = 0; ni < 2; ++ni) {
            int gc = nBase + wn + ni * 16 + col;
            float bv = BIAS ? bias[gc] : 0.0f;
            #pragma unroll
            for (int r = 0; r < 8; ++r) {
                int gr = mBase + wm + mi * 16 + rowOff + r;
                C[(size_t)gr * Nout + gc] = (TC)(acc[mi][ni][r] + bv);
            }
        }
}

// ---------------------------------------------------------------------------
// WvS[d,i] = sum_h Wv[h*64+d, i]  (f32 in, f16 out, 64x1024)
// ---------------------------------------------------------------------------
__global__ __launch_bounds__(256) void wv_headsum_kernel(
    const float* __restrict__ Wv, _Float16* __restrict__ WvS)
{
    int idx = blockIdx.x * 256 + threadIdx.x;      // 64*1024 total
    int d = idx >> 10, i = idx & 1023;
    float s = 0.f;
    #pragma unroll
    for (int h = 0; h < Hh; ++h)
        s += Wv[(size_t)(h * Dd + d) * Ed + i];
    WvS[idx] = (_Float16)s;
}

// ---------------------------------------------------------------------------
// energy[n,qh,kh,s] = sum_d Qh[tok, qh*64+d] * Kh[tok, kh*64+d]
// One wave per token: 16x16 output, K=64 => 2 WMMA k-steps.
// ---------------------------------------------------------------------------
__global__ __launch_bounds__(256) void energy_kernel(
    const _Float16* __restrict__ Q, const _Float16* __restrict__ Kh,
    float* __restrict__ energy)
{
    const int lane  = threadIdx.x & 31;
    const int wave  = threadIdx.x >> 5;
    const int token = blockIdx.x * 8 + wave;       // 16384 tokens exactly
    const _Float16* qrow = Q  + (size_t)token * Ed;
    const _Float16* krow = Kh + (size_t)token * Ed;
    const int fr  = lane & 15;
    const int kgA = (lane >> 4) * 8;
    const int kgB = (lane >> 4) * 8;

    v8f acc = {};
    #pragma unroll
    for (int ks = 0; ks < 2; ++ks) {
        const int d0 = ks * 32;
        v16h aF = make_frag(qrow + fr * Dd + d0 + kgA,
                            qrow + fr * Dd + d0 + kgA + 16);
        v16h bF = make_frag(krow + fr * Dd + d0 + kgB,
                            krow + fr * Dd + d0 + kgB + 16);
        acc = __builtin_amdgcn_wmma_f32_16x16x32_f16(
            false, aF, false, bF, (short)0, acc, false, false);
    }

    const int n = token >> 12, s = token & 4095;
    const int khc = lane & 15;
    const int qh0 = (lane >> 4) * 8;
    #pragma unroll
    for (int r = 0; r < 8; ++r) {
        int qh = qh0 + r;
        energy[(((size_t)n * Hh + qh) * Hh + khc) * Sq + s] = acc[r];
    }
}

// ---------------------------------------------------------------------------
// Per (n,qh,kh) row of 4096: m = max(e/8), Z = sum exp(e/8 - m)
// ---------------------------------------------------------------------------
__global__ __launch_bounds__(256) void softmax_stats_kernel(
    const float* __restrict__ energy, float2* __restrict__ stats)
{
    __shared__ float red[256];
    const int row = blockIdx.x;                    // 1024 rows
    const int tid = threadIdx.x;
    const float* e = energy + (size_t)row * Sq;

    float m = -3.4e38f;
    for (int i = tid; i < Sq; i += 256) m = fmaxf(m, e[i] * 0.125f);
    red[tid] = m; __syncthreads();
    for (int s2 = 128; s2 > 0; s2 >>= 1) {
        if (tid < s2) red[tid] = fmaxf(red[tid], red[tid + s2]);
        __syncthreads();
    }
    m = red[0]; __syncthreads();

    float sum = 0.f;
    for (int i = tid; i < Sq; i += 256) sum += __expf(e[i] * 0.125f - m);
    red[tid] = sum; __syncthreads();
    for (int s2 = 128; s2 > 0; s2 >>= 1) {
        if (tid < s2) red[tid] += red[tid + s2];
        __syncthreads();
    }
    if (tid == 0) stats[row] = make_float2(m, red[0]);
}

// ---------------------------------------------------------------------------
// A[n,qh,s] = sum_kh exp(e[n,qh,kh,s]/8 - m)/Z
// ---------------------------------------------------------------------------
__global__ __launch_bounds__(256) void attn_rowsum_kernel(
    const float* __restrict__ energy, const float2* __restrict__ stats,
    float* __restrict__ A)
{
    const int nq = blockIdx.x >> 4;                         // n*16+qh (64)
    const int s  = ((blockIdx.x & 15) << 8) + threadIdx.x;  // 0..4095
    float acc = 0.f;
    #pragma unroll
    for (int kh = 0; kh < Hh; ++kh) {
        float2 st = stats[nq * Hh + kh];
        float e = energy[((size_t)(nq * Hh + kh)) * Sq + s];
        acc += __expf(e * 0.125f - st.x) / st.y;
    }
    A[(size_t)nq * Sq + s] = acc;
}

// ---------------------------------------------------------------------------
// Y[n, qh*256 + s/16, (s%16)*64 + d] = f16(A[n,qh,s] * Vs[n,s,d])
// (the reference's (N,H,S,D)->(N,S,E) reshape)
// ---------------------------------------------------------------------------
__global__ __launch_bounds__(256) void build_y_kernel(
    const float* __restrict__ A, const float* __restrict__ Vs,
    _Float16* __restrict__ Y)
{
    size_t idx = (size_t)blockIdx.x * 256 + threadIdx.x;   // 16,777,216
    int d  = (int)(idx & 63);
    int s  = (int)((idx >> 6) & 4095);
    int qh = (int)((idx >> 18) & 15);
    int n  = (int)(idx >> 22);
    float a  = A[((size_t)n * Hh + qh) * Sq + s];
    float vs = Vs[((size_t)n * Sq + s) * Dd + d];
    int srow = qh * 256 + (s >> 4);
    int ecol = (s & 15) * Dd + d;
    Y[((size_t)n * Sq + srow) * Ed + ecol] = (_Float16)(a * vs);
}

// ---------------------------------------------------------------------------
extern "C" void kernel_launch(void* const* d_in, const int* in_sizes, int n_in,
                              void* d_out, int out_size, void* d_ws, size_t ws_size,
                              hipStream_t stream)
{
    const float* x  = (const float*)d_in[0];
    const float* Wq = (const float*)d_in[1];
    const float* Wk = (const float*)d_in[2];
    const float* Wv = (const float*)d_in[3];
    const float* Wo = (const float*)d_in[4];
    const float* bo = (const float*)d_in[5];
    float* out = (float*)d_out;

    const int M = Nb * Sq;                 // 16384 token rows
    const size_t EE = (size_t)Ed * Ed;     // 1,048,576

    // Workspace carve-up (~155 MB total)
    char* ws = (char*)d_ws;
    size_t off = 0;
    auto carve = [&](size_t bytes) -> void* {
        void* p = ws + off;
        off = (off + bytes + 255) & ~(size_t)255;
        return p;
    };
    _Float16* hx    = (_Float16*)carve((size_t)M * Ed * 2);        // 32 MB
    _Float16* hWq   = (_Float16*)carve(EE * 2);                    //  2 MB
    _Float16* hWk   = (_Float16*)carve(EE * 2);                    //  2 MB
    _Float16* hWo   = (_Float16*)carve(EE * 2);                    //  2 MB
    _Float16* Qh    = (_Float16*)carve((size_t)M * Ed * 2);        // 32 MB
    _Float16* Kh    = (_Float16*)carve((size_t)M * Ed * 2);        // 32 MB
    _Float16* Yb    = (_Float16*)carve((size_t)M * Ed * 2);        // 32 MB
    float*    energy= (float*)   carve((size_t)Nb*Hh*Hh*Sq * 4);   // 16 MB
    float*    Vs    = (float*)   carve((size_t)M * Dd * 4);        //  4 MB
    float*    Abuf  = (float*)   carve((size_t)Nb*Hh*Sq * 4);      //  1 MB
    _Float16* WvS   = (_Float16*)carve((size_t)Dd * Ed * 2);       // 128 KB
    float2*   stats = (float2*)  carve((size_t)Nb*Hh*Hh * 8);      //  8 KB

    dim3 blk(256);
    dim3 gemmGrid(M / 128, Ed / 64);       // 128 x 16

    // f16 working copies (one pass each)
    f32_to_f16_kernel<<<dim3((size_t)M * Ed / 8 / 256), blk, 0, stream>>>(x, hx);
    f32_to_f16_kernel<<<dim3(EE / 8 / 256), blk, 0, stream>>>(Wq, hWq);
    f32_to_f16_kernel<<<dim3(EE / 8 / 256), blk, 0, stream>>>(Wk, hWk);
    f32_to_f16_kernel<<<dim3(EE / 8 / 256), blk, 0, stream>>>(Wo, hWo);

    // Q = x @ Wq.T, K = x @ Wk.T  (f16 outputs)
    wmma_gemm_kernel<_Float16, false>
        <<<gemmGrid, blk, 0, stream>>>(hx, hWq, Qh, nullptr, M, Ed, Ed);
    wmma_gemm_kernel<_Float16, false>
        <<<gemmGrid, blk, 0, stream>>>(hx, hWk, Kh, nullptr, M, Ed, Ed);

    // WvS = head-sum of Wv;  Vs = x @ WvS.T  (f32 output, Nout=64)
    wv_headsum_kernel<<<dim3((Dd * Ed) / 256), blk, 0, stream>>>(Wv, WvS);
    wmma_gemm_kernel<float, false>
        <<<dim3(M / 128, 1), blk, 0, stream>>>(hx, WvS, Vs, nullptr, M, Dd, Ed);

    // energy (WMMA, one wave per token)
    energy_kernel<<<dim3(M / 8), blk, 0, stream>>>(Qh, Kh, energy);

    // softmax stats + row-sum of attention over kh
    softmax_stats_kernel<<<dim3(Nb * Hh * Hh), blk, 0, stream>>>(energy, stats);
    attn_rowsum_kernel<<<dim3(Nb * Hh * 16), blk, 0, stream>>>(energy, stats, Abuf);

    // Y = outer(A, Vs) in the reference's reshuffled (N,S,E) layout
    build_y_kernel<<<dim3((size_t)M * Ed / 256), blk, 0, stream>>>(Abuf, Vs, Yb);

    // out = Y @ Wo.T + bo
    wmma_gemm_kernel<float, true>
        <<<gemmGrid, blk, 0, stream>>>(Yb, hWo, out, bo, M, Ed, Ed);
}
